// Attention_19834158973667
// MI455X (gfx1250) — matrix operand order repaired and checked by reading, not compiled, not run on hardware
//
#include <hip/hip_runtime.h>

// ---------------------------------------------------------------------------
// Spatial-reduction attention (PVT-style) for MI455X / gfx1250, wave32 WMMA.
// B=4, N=4096 (64x64), C=256, heads=8, d=32, SR=2 -> Nk=1024.
// ---------------------------------------------------------------------------

typedef _Float16 f16;
typedef __attribute__((ext_vector_type(16))) _Float16 v16h;
typedef __attribute__((ext_vector_type(8)))  _Float16 v8h;
typedef __attribute__((ext_vector_type(8)))  float    v8f;

#define HEADS     8
#define HDIM      32
#define NTOK      4096      // 64*64 query tokens
#define NKTOK     1024      // 32*32 kv tokens
#define CDIM      256
#define ATT_SCALE 0.17677669529663687f  // 32^-0.5
#define LN_EPS    1e-5f

// ---------------------------------------------------------------------------
// WMMA fragment loaders (16x16x32 f16, contiguous k order used here; A and B share it)
// A (16xK row-major): lane L -> row M=L%16; element e -> K = 16*hg+e
//   => two contiguous 8-half (16B) chunks at k0+16*hg and k0+16*hg+8.
// B as Bt[N][K] (K contiguous): lane L -> col N=L%16; element e -> K=16*hg+e
//   => one contiguous 16-half (32B) chunk at k0+16*hg.
// D f32: lane L holds (m = r+8*(L/16), n = L%16) in vgpr r.
// ---------------------------------------------------------------------------
__device__ __forceinline__ v16h load_a_frag(const f16* __restrict__ A, int lda,
                                            int row, int k0, int hg) {
  const f16* p = A + (size_t)row * lda + k0 + 16 * hg;
  v8h lo = *(const v8h*)(p);
  v8h hi = *(const v8h*)(p + 8);
  v16h a;
#pragma unroll
  for (int i = 0; i < 8; ++i) { a[i] = lo[i]; a[i + 8] = hi[i]; }
  return a;
}

#define WMMA_F16(A_, B_, C_)                                                   \
  __builtin_amdgcn_wmma_f32_16x16x32_f16(false, (A_), false, (B_), (short)0,   \
                                         (C_), false, false)

// ---------------------------------------------------------------------------
// WMMA GEMM, one wave per 16x64 output strip (4 accumulators, A-frag reused
// 4x per k-step), software-pipelined k-loop (next-step fragments loaded while
// current WMMAs execute).  C[M,N] = A[M,K] * Bt[N,K]^T + bias.   N % 64 == 0.
// ---------------------------------------------------------------------------
__global__ __launch_bounds__(32) void wmma_gemm_bias(
    const f16* __restrict__ A, const f16* __restrict__ Bt,
    const float* __restrict__ bias, float* __restrict__ C,
    int M, int N, int K) {
  const int nnb  = N >> 6;                 // 64-wide strips
  const int tm   = blockIdx.x / nnb;
  const int tnb  = blockIdx.x % nnb;
  const int lane = threadIdx.x;
  const int lr   = lane & 15;
  const int hg   = lane >> 4;
  const int arow = tm * 16 + lr;
  const int brow = tnb * 64 + lr;          // + j*16 per sub-tile

  const f16* bp0 = Bt + (size_t)(brow     ) * K + 16 * hg;
  const f16* bp1 = Bt + (size_t)(brow + 16) * K + 16 * hg;
  const f16* bp2 = Bt + (size_t)(brow + 32) * K + 16 * hg;
  const f16* bp3 = Bt + (size_t)(brow + 48) * K + 16 * hg;

  v8f acc0 = {}, acc1 = {}, acc2 = {}, acc3 = {};

  // prologue: fragments for k0 = 0
  v16h a  = load_a_frag(A, K, arow, 0, hg);
  v16h b0 = *(const v16h*)(bp0);
  v16h b1 = *(const v16h*)(bp1);
  v16h b2 = *(const v16h*)(bp2);
  v16h b3 = *(const v16h*)(bp3);

  for (int k0 = 32; k0 < K; k0 += 32) {
    if (k0 + 96 < K) {                     // stay inside the row / buffer
      __builtin_prefetch(A + (size_t)arow * K + k0 + 96, 0, 3);
      __builtin_prefetch(bp0 + k0 + 96, 0, 3);
    }
    // issue next-step loads into fresh registers...
    v16h an  = load_a_frag(A, K, arow, k0, hg);
    v16h b0n = *(const v16h*)(bp0 + k0);
    v16h b1n = *(const v16h*)(bp1 + k0);
    v16h b2n = *(const v16h*)(bp2 + k0);
    v16h b3n = *(const v16h*)(bp3 + k0);
    // ...while the current step's WMMAs execute
    acc0 = WMMA_F16(a, b0, acc0);
    acc1 = WMMA_F16(a, b1, acc1);
    acc2 = WMMA_F16(a, b2, acc2);
    acc3 = WMMA_F16(a, b3, acc3);
    a = an; b0 = b0n; b1 = b1n; b2 = b2n; b3 = b3n;
  }
  // epilogue: last k-step
  acc0 = WMMA_F16(a, b0, acc0);
  acc1 = WMMA_F16(a, b1, acc1);
  acc2 = WMMA_F16(a, b2, acc2);
  acc3 = WMMA_F16(a, b3, acc3);

  const v8f accs[4] = {acc0, acc1, acc2, acc3};
#pragma unroll
  for (int j = 0; j < 4; ++j) {
    const int n  = tnb * 64 + j * 16 + lr;
    const float bv = bias ? bias[n] : 0.0f;
#pragma unroll
    for (int r = 0; r < 8; ++r) {
      const int m = tm * 16 + 8 * hg + r;
      C[(size_t)m * N + n] = accs[j][r] + bv;
    }
  }
}

// ---------------------------------------------------------------------------
// f32 -> f16 cast
// ---------------------------------------------------------------------------
__global__ void cast_f32_f16(const float* __restrict__ in, f16* __restrict__ out,
                             int n) {
  int i = blockIdx.x * blockDim.x + threadIdx.x;
  if (i < n) out[i] = (f16)in[i];
}

// ---------------------------------------------------------------------------
// Depthwise 3x3 'SAME' conv on x[B,N,C] (64x64 grid) -> q_f16[(b*8+h)][n][d]
// ---------------------------------------------------------------------------
__global__ void qconv_kernel(const float* __restrict__ x,
                             const float* __restrict__ w,
                             f16* __restrict__ q, int total) {
  int idx = blockIdx.x * blockDim.x + threadIdx.x;
  if (idx >= total) return;
  const int xp = idx & 63, y = (idx >> 6) & 63, c = (idx >> 12) & 255,
            b  = idx >> 20;
  float s = 0.0f;
#pragma unroll
  for (int kh = 0; kh < 3; ++kh) {
    const int yy = y + kh - 1;
    if (yy < 0 || yy >= 64) continue;
#pragma unroll
    for (int kw = 0; kw < 3; ++kw) {
      const int xx = xp + kw - 1;
      if (xx < 0 || xx >= 64) continue;
      s += w[c * 9 + kh * 3 + kw] *
           x[((size_t)b * NTOK + yy * 64 + xx) * CDIM + c];
    }
  }
  const int head = c >> 5, d = c & 31, n = y * 64 + xp;
  q[(((size_t)(b * HEADS + head)) * NTOK + n) * HDIM + d] = (f16)s;
}

// ---------------------------------------------------------------------------
// im2col for the SR conv (kernel==stride==2 -> pure permutation of x), in f16.
// P[b*1024 + hs*32+ws][ci*4 + kh*2 + kw] = x[b][(2hs+kh)*64 + 2ws+kw][ci]
// ---------------------------------------------------------------------------
__global__ void im2col_sr(const float* __restrict__ x, f16* __restrict__ P,
                          int total) {
  int idx = blockIdx.x * blockDim.x + threadIdx.x;
  if (idx >= total) return;
  const int col = idx & 1023;
  const int row = idx >> 10;
  const int b = row >> 10, nk = row & 1023;
  const int hs = nk >> 5, ws = nk & 31;
  const int ci = col >> 2, kh = (col >> 1) & 1, kw = col & 1;
  const int n = (2 * hs + kh) * 64 + (2 * ws + kw);
  P[idx] = (f16)x[((size_t)b * NTOK + n) * CDIM + ci];
}

// ---------------------------------------------------------------------------
// In-place LayerNorm over C=256, one wave per row, 8 elems/lane.
// ---------------------------------------------------------------------------
__global__ __launch_bounds__(32) void layernorm_rows(float* __restrict__ xr,
                                                     const float* __restrict__ g,
                                                     const float* __restrict__ bta) {
  const int row = blockIdx.x;
  const int lane = threadIdx.x;
  float v[8];
  float s = 0.0f;
#pragma unroll
  for (int i = 0; i < 8; ++i) {
    v[i] = xr[(size_t)row * CDIM + lane + i * 32];
    s += v[i];
  }
#pragma unroll
  for (int off = 16; off > 0; off >>= 1) s += __shfl_xor(s, off, 32);
  const float mu = s * (1.0f / CDIM);
  float vs = 0.0f;
#pragma unroll
  for (int i = 0; i < 8; ++i) {
    const float d = v[i] - mu;
    vs += d * d;
  }
#pragma unroll
  for (int off = 16; off > 0; off >>= 1) vs += __shfl_xor(vs, off, 32);
  const float inv = rsqrtf(vs * (1.0f / CDIM) + LN_EPS);
#pragma unroll
  for (int i = 0; i < 8; ++i) {
    const int c = lane + i * 32;
    xr[(size_t)row * CDIM + c] = (v[i] - mu) * inv * g[c] + bta[c];
  }
}

// ---------------------------------------------------------------------------
// Depthwise 3x3 'SAME' conv on xr[B,Nk,C] (32x32 grid), 512 out channels,
// JAX/torch group-major: out channel o reads input channel o/2, kernel w[o].
// k = out ch 0..255 -> k_f16[(b*8+h)][nk][d]  (B-operand of QK^T)
// v = out ch 256..511 -> vT_f16[(b*8+h)][d][nk] (B-operand of PV)
// ---------------------------------------------------------------------------
__global__ void kvconv_kernel(const float* __restrict__ xr,
                              const float* __restrict__ w,
                              f16* __restrict__ kf, f16* __restrict__ vT,
                              int total) {
  int idx = blockIdx.x * blockDim.x + threadIdx.x;
  if (idx >= total) return;
  const int nk = idx & 1023;
  const int o  = (idx >> 10) & 511;
  const int b  = idx >> 19;
  const int hs = nk >> 5, ws = nk & 31;
  const int gch = o >> 1;  // input channel
  float s = 0.0f;
#pragma unroll
  for (int kh = 0; kh < 3; ++kh) {
    const int yy = hs + kh - 1;
    if (yy < 0 || yy >= 32) continue;
#pragma unroll
    for (int kw = 0; kw < 3; ++kw) {
      const int xx = ws + kw - 1;
      if (xx < 0 || xx >= 32) continue;
      s += w[o * 9 + kh * 3 + kw] *
           xr[((size_t)b * NKTOK + yy * 32 + xx) * CDIM + gch];
    }
  }
  if (o < 256) {
    const int head = o >> 5, d = o & 31;
    kf[(((size_t)(b * HEADS + head)) * NKTOK + nk) * HDIM + d] = (f16)s;
  } else {
    const int c = o - 256;
    const int head = c >> 5, d = c & 31;
    vT[(((size_t)(b * HEADS + head)) * HDIM + d) * NKTOK + nk] = (f16)s;
  }
}

// ---------------------------------------------------------------------------
// Fused attention: one wave handles 16 query rows of one (b,head).
// 64 QK^T WMMAs -> f16 scores in LDS -> f32 softmax (all 32 lanes, in-place
// exp, 1/sum folded into the epilogue) -> 64 PV WMMAs -> f16 out.
// Score row stride 1048 halves = 524 dwords; 524 % 64 = 12 and gcd(12,64)=4
// => the 16 rows hit 16 distinct banks during column sweeps; the paired
// half-wave lanes read interleaved columns of the same dword (broadcast).
// Global-fed loops are software-pipelined like the GEMM.
// ---------------------------------------------------------------------------
#define SROWH 1048

__global__ __launch_bounds__(32) void attn_kernel(const f16* __restrict__ q,
                                                  const f16* __restrict__ kf,
                                                  const f16* __restrict__ vT,
                                                  f16* __restrict__ outp) {
  __shared__ f16   sc[16 * SROWH];  // scores, then probabilities (33536 B)
  __shared__ float invs[16];

  const int wid  = blockIdx.x;
  const int mt   = wid & 255;  // 256 query tiles of 16 rows
  const int bh   = wid >> 8;   // b*8 + head
  const int lane = threadIdx.x;
  const int lr   = lane & 15;
  const int hg   = lane >> 4;

  const f16* qb = q  + (size_t)bh * NTOK  * HDIM;
  const f16* kb = kf + (size_t)bh * NKTOK * HDIM + (size_t)lr * HDIM + 16 * hg;
  const f16* vb = vT + (size_t)bh * HDIM  * NKTOK;

  // Q fragment (K = d = 32, single frag), reused for all 64 score tiles.
  const v16h aq = load_a_frag(qb, HDIM, mt * 16 + lr, 0, hg);

  // ---- scores: S[16,1024] = scale * Q K^T  (pipelined K-frag loads) ----
  {
    v16h bk = *(const v16h*)(kb);
    for (int n0 = 0; n0 < NKTOK; n0 += 16) {
      v16h bkn = bk;
      if (n0 + 16 < NKTOK)
        bkn = *(const v16h*)(kb + (size_t)(n0 + 16) * HDIM);
      v8f s = {};
      s = WMMA_F16(aq, bk, s);
#pragma unroll
      for (int r = 0; r < 8; ++r)
        sc[(r + 8 * hg) * SROWH + n0 + lr] = (f16)(s[r] * ATT_SCALE);
      bk = bkn;
    }
  }
  __syncthreads();

  // ---- softmax: row lr split across the lane pair (lr, lr+16); columns are
  // interleaved (j = 2t + hg) so both lanes read the same LDS dword. ----
  {
    const int rbase = lr * SROWH;
    float mx = -3.0e38f;
    for (int t = 0; t < NKTOK / 2; ++t)
      mx = fmaxf(mx, (float)sc[rbase + 2 * t + hg]);
    mx = fmaxf(mx, __shfl_xor(mx, 16, 32));
    float sum = 0.0f;
    for (int t = 0; t < NKTOK / 2; ++t) {
      const int jj = rbase + 2 * t + hg;
      const float p = __expf((float)sc[jj] - mx);
      sum += p;
      sc[jj] = (f16)p;
    }
    sum += __shfl_xor(sum, 16, 32);
    if (hg == 0) invs[lr] = 1.0f / sum;
  }
  __syncthreads();

  // ---- out[16,32] = P V : two 16-wide d tiles, K = Nk = 1024 ----
  v8f acc0 = {}, acc1 = {};
  {
    const f16* vb0 = vb + (size_t)lr        * NKTOK + 16 * hg;
    const f16* vb1 = vb + (size_t)(16 + lr) * NKTOK + 16 * hg;
    v16h b0 = *(const v16h*)(vb0);
    v16h b1 = *(const v16h*)(vb1);
    for (int k0 = 0; k0 < NKTOK; k0 += 32) {
      v16h b0n = b0, b1n = b1;
      if (k0 + 32 < NKTOK) {
        b0n = *(const v16h*)(vb0 + k0 + 32);
        b1n = *(const v16h*)(vb1 + k0 + 32);
      }
      v16h ap;
#pragma unroll
      for (int i = 0; i < 8; ++i) {
        ap[i]     = sc[lr * SROWH + k0 + 16 * hg + i];
        ap[i + 8] = sc[lr * SROWH + k0 + 16 * hg + 8 + i];
      }
      acc0 = WMMA_F16(ap, b0, acc0);
      acc1 = WMMA_F16(ap, b1, acc1);
      b0 = b0n; b1 = b1n;
    }
  }

  const int b  = bh >> 3;
  const int c0 = (bh & 7) * HDIM;  // head * 32
  const size_t rowbase = (size_t)b * NTOK + mt * 16;
#pragma unroll
  for (int r = 0; r < 8; ++r) {
    const int m = r + 8 * hg;
    const float iv = invs[m];
    outp[(rowbase + m) * CDIM + c0 + lr]      = (f16)(acc0[r] * iv);
    outp[(rowbase + m) * CDIM + c0 + 16 + lr] = (f16)(acc1[r] * iv);
  }
}

// ---------------------------------------------------------------------------
// Host launcher
// ---------------------------------------------------------------------------
extern "C" void kernel_launch(void* const* d_in, const int* in_sizes, int n_in,
                              void* d_out, int out_size, void* d_ws,
                              size_t ws_size, hipStream_t stream) {
  (void)n_in; (void)out_size; (void)ws_size;
  const float* x      = (const float*)d_in[0];
  const float* to_q_w = (const float*)d_in[1];
  const float* to_kv_w= (const float*)d_in[2];
  const float* sr_w   = (const float*)d_in[3];
  const float* sr_b   = (const float*)d_in[4];
  const float* ln_g   = (const float*)d_in[5];
  const float* ln_b   = (const float*)d_in[6];
  const float* proj_w = (const float*)d_in[7];
  const float* proj_b = (const float*)d_in[8];

  const int B = in_sizes[0] / (NTOK * CDIM);  // 4

  // ---- workspace carving (bytes, 256-aligned sizes) ----
  char* ws = (char*)d_ws;
  size_t off = 0;
  f16* q_f16 = (f16*)(ws + off); off += (size_t)B * HEADS * NTOK * HDIM * 2;   // 8 MB
  f16* P     = (f16*)(ws + off); off += (size_t)B * NKTOK * 1024 * 2;          // 8 MB
  f16* Wsr   = (f16*)(ws + off); off += (size_t)CDIM * 1024 * 2;               // 512 KB
  f16* Wp    = (f16*)(ws + off); off += (size_t)CDIM * CDIM * 2;               // 128 KB
  float* xr  = (float*)(ws + off); off += (size_t)B * NKTOK * CDIM * 4;        // 4 MB
  f16* kf    = (f16*)(ws + off); off += (size_t)B * HEADS * NKTOK * HDIM * 2;  // 2 MB
  f16* vT    = (f16*)(ws + off); off += (size_t)B * HEADS * HDIM * NKTOK * 2;  // 2 MB
  f16* ao    = (f16*)(ws + off); off += (size_t)B * NTOK * CDIM * 2;           // 8 MB

  // 1) weight casts to f16 (flat layouts are already the Bt[N][K] operand)
  cast_f32_f16<<<(CDIM * 1024 + 255) / 256, 256, 0, stream>>>(sr_w, Wsr,
                                                              CDIM * 1024);
  cast_f32_f16<<<(CDIM * CDIM + 255) / 256, 256, 0, stream>>>(proj_w, Wp,
                                                              CDIM * CDIM);
  // 2) Q depthwise conv
  {
    int total = B * CDIM * NTOK;
    qconv_kernel<<<(total + 255) / 256, 256, 0, stream>>>(x, to_q_w, q_f16,
                                                          total);
  }
  // 3) SR conv: im2col permutation + WMMA GEMM [B*1024,1024]x[1024,256]+bias
  {
    int total = B * NKTOK * 1024;
    im2col_sr<<<(total + 255) / 256, 256, 0, stream>>>(x, P, total);
    int M = B * NKTOK;
    wmma_gemm_bias<<<(M / 16) * (CDIM / 64), 32, 0, stream>>>(
        P, Wsr, sr_b, xr, M, CDIM, 1024);
  }
  // 4) LayerNorm (in place)
  layernorm_rows<<<B * NKTOK, 32, 0, stream>>>(xr, ln_g, ln_b);
  // 5) KV depthwise conv
  {
    int total = B * 512 * NKTOK;
    kvconv_kernel<<<(total + 255) / 256, 256, 0, stream>>>(xr, to_kv_w, kf, vT,
                                                           total);
  }
  // 6) fused attention (one wave per 16 query rows per (b,head))
  attn_kernel<<<B * HEADS * (NTOK / 16), 32, 0, stream>>>(q_f16, kf, vT, ao);
  // 7) output projection: [B*4096,256]x[256,256] + bias -> fp32 d_out
  {
    int M = B * NTOK;
    wmma_gemm_bias<<<(M / 16) * (CDIM / 64), 32, 0, stream>>>(
        ao, Wp, proj_b, (float*)d_out, M, CDIM, CDIM);
  }
}
